// NoGNNMLPPredictor_35390530519867
// MI455X (gfx1250) — matrix operand order repaired and checked by reading, not compiled, not run on hardware
//
#include <hip/hip_runtime.h>

// ---- types for WMMA fragments (wave32, gfx1250) ----
typedef __attribute__((ext_vector_type(16))) _Float16 v16h;
typedef __attribute__((ext_vector_type(8)))  _Float16 v8h;
typedef __attribute__((ext_vector_type(2)))  _Float16 v2h;
typedef __attribute__((ext_vector_type(8)))  float    v8f;
typedef __attribute__((ext_vector_type(4)))  float    v4f;
typedef __attribute__((ext_vector_type(2)))  float    v2f;

#define PITCH1 136   // padded LDS row pitch (halves): 68 dword banks/row -> conflict-free
#define PITCH2 72
#define THREADS 256
#define WAVES   (THREADS / 32)

// ---------------------------------------------------------------------------
// Operand convention (all layers): A = WEIGHTS, B = ACTIVATIONS.
//   D[o][e] = sum_k W[o][k] * x[e][k]   (o on M/registers, e on N/lanes)
// D keeps edges on lanes -> next layer's B operand needs only an XOR-16
// half-swap between lane-halves (in registers), no LDS bounce.
// ---------------------------------------------------------------------------

// Weight A fragment from padded LDS (two contiguous 8-half runs per lane).
__device__ __forceinline__ v16h wmma_a_frag_lds(const _Float16* X, int m, int pitch,
                                                int kstep, int lane) {
  const int kh = (lane >> 4) & 1;
  const _Float16* row = X + m * pitch + kstep * 32;
  v8h r0 = *(const v8h*)(row + 16 * kh);
  v8h r1 = *(const v8h*)(row + 16 * kh + 8);
  v16h r;
#pragma unroll
  for (int j = 0; j < 8; ++j) { r[j] = r0[j]; r[j + 8] = r1[j]; }
  return r;
}

// Gather he = [h[s], h[d]] as 4 B fragments from f32 h (load + cvt_pk).
__device__ __forceinline__ void gather_heB_f32(const float* __restrict__ h, int s, int d,
                                               int khi, v16h b[4]) {
  const float* ps = h + (size_t)s * 64 + 16 * khi;
  const float* pd = h + (size_t)d * 64 + 16 * khi;
  v4f raw[16];
#pragma unroll
  for (int half = 0; half < 2; ++half) {
#pragma unroll
    for (int q = 0; q < 4; ++q) {
      raw[half * 4 + q]     = *(const v4f*)(ps + 32 * half + 4 * q);
      raw[8 + half * 4 + q] = *(const v4f*)(pd + 32 * half + 4 * q);
    }
  }
#pragma unroll
  for (int ks = 0; ks < 4; ++ks) {
    v16h f;
#pragma unroll
    for (int q = 0; q < 4; ++q) {
      v4f r4 = raw[(ks < 2 ? 0 : 8) + (ks & 1) * 4 + q];
#pragma unroll
      for (int j = 0; j < 4; ++j) f[q * 4 + j] = (_Float16)r4[j];
    }
    b[ks] = f;
  }
}

// Gather from pre-converted f16 h: fragments load directly, zero conversion.
__device__ __forceinline__ void gather_heB_f16(const _Float16* __restrict__ h16,
                                               int s, int d, int khi, v16h b[4]) {
  const _Float16* ps = h16 + (size_t)s * 64 + 16 * khi;
  const _Float16* pd = h16 + (size_t)d * 64 + 16 * khi;
#pragma unroll
  for (int ks = 0; ks < 4; ++ks) {
    const _Float16* p = (ks < 2 ? ps : pd) + 32 * (ks & 1);
    v8h lo = *(const v8h*)p;
    v8h hi = *(const v8h*)(p + 8);
    v16h f;
#pragma unroll
    for (int j = 0; j < 8; ++j) { f[j] = lo[j]; f[j + 8] = hi[j]; }
    b[ks] = f;
  }
}

// Pack 8 f32 accumulators into 4 packed-f16 dwords, ReLU applied in f16.
__device__ __forceinline__ void pack_relu(const v8f acc, int pk[4]) {
  const v2h z = {};
#pragma unroll
  for (int j = 0; j < 4; ++j) {
    v2h p; p[0] = (_Float16)acc[2 * j]; p[1] = (_Float16)acc[2 * j + 1];
    v2h m = __builtin_elementwise_max(p, z);
    pk[j] = __builtin_bit_cast(int, m);
  }
}

// Build next layer's B fragment for K-block [32k, 32k+32) from packed outputs
// of channel groups gA=2k and gB=2k+1 via XOR-16 half-swap.
__device__ __forceinline__ v16h build_bfrag_swap(const int* pkA, const int* pkB, int khi) {
  v16h f;
#pragma unroll
  for (int j = 0; j < 4; ++j) {
    const int send = khi ? pkA[j] : pkB[j];
    const int recv = __shfl_xor(send, 16, 32);
    const int lo = khi ? recv : pkA[j];
    const int hi = khi ? pkB[j] : recv;
    const v2h l2 = __builtin_bit_cast(v2h, lo);
    const v2h h2 = __builtin_bit_cast(v2h, hi);
    f[2 * j]     = l2[0]; f[2 * j + 1]     = l2[1];
    f[8 + 2 * j] = h2[0]; f[8 + 2 * j + 1] = h2[1];
  }
  return f;
}

// Per-register bias vector for C init: acc[r] = bias[16g + 8*khi + r].
__device__ __forceinline__ v8f load_bias8(const float* bptr) {
  v4f a = *(const v4f*)bptr;
  v4f b = *(const v4f*)(bptr + 4);
  v8f c;
#pragma unroll
  for (int j = 0; j < 4; ++j) { c[j] = a[j]; c[4 + j] = b[j]; }
  return c;
}

__device__ __forceinline__ void load_idx(const int* __restrict__ src,
                                         const int* __restrict__ dst,
                                         int iter, int n, int E, int* ss, int* dd) {
  const int b = iter * 32;
#pragma unroll
  for (int t = 0; t < 2; ++t) {
    int e = b + 16 * t + n;
    if (e >= E) e = E - 1;
    ss[t] = src[e];
    dd[t] = dst[e];
  }
}

// One-time f32 -> f16 conversion of the node-feature table into workspace.
__global__ void h_to_f16(const float* __restrict__ hf, _Float16* __restrict__ h16,
                         int total8) {
  const int i = blockIdx.x * blockDim.x + threadIdx.x;
  if (i >= total8) return;
  const float* p = hf + (size_t)i * 8;
  v4f a = *(const v4f*)p;
  v4f b = *(const v4f*)(p + 4);
  v8h o;
#pragma unroll
  for (int j = 0; j < 4; ++j) { o[j] = (_Float16)a[j]; o[4 + j] = (_Float16)b[j]; }
  *(v8h*)(h16 + (size_t)i * 8) = o;
}

template <bool USE16>
__global__ __launch_bounds__(THREADS)
void edge_mlp_wmma(const float* __restrict__ h,
                   const _Float16* __restrict__ h16,
                   const int* __restrict__ src,
                   const int* __restrict__ dst,
                   const float* __restrict__ Win_w,
                   const float* __restrict__ Win_b,
                   const float* __restrict__ W1_w,
                   const float* __restrict__ W1_b,
                   const float* __restrict__ W2_w,
                   const float* __restrict__ W2_b,
                   float* __restrict__ out,
                   int E) {
  extern __shared__ _Float16 smem[];
  _Float16* Win16 = smem;                       // 128 x PITCH1
  _Float16* W116  = Win16 + 128 * PITCH1;       // 64 x PITCH1
  _Float16* W2p   = W116  + 64 * PITCH1;        // 16 x PITCH2 (rows >=2 zero)

  const int tid = threadIdx.x;

  // ---- one-time per-block weight staging: f32 -> f16 into padded LDS ----
  for (int i = tid; i < 128 * 128; i += blockDim.x)
    Win16[(i >> 7) * PITCH1 + (i & 127)] = (_Float16)Win_w[i];
  for (int i = tid; i < 64 * 128; i += blockDim.x)
    W116[(i >> 7) * PITCH1 + (i & 127)] = (_Float16)W1_w[i];
  for (int i = tid; i < 16 * 64; i += blockDim.x) {
    const int o = i >> 6, c = i & 63;
    W2p[o * PITCH2 + c] = (o < 2) ? (_Float16)W2_w[o * 64 + c] : (_Float16)0.0f;
  }
  __syncthreads();

  const int lane = tid & 31;
  const int n    = lane & 15;
  const int khi  = lane >> 4;

  // ---- loop-invariant: layer-3 weight A fragments + output bias.
  // Layer-2 weights (W116) and bias (W1_b) are intentionally NOT laundered:
  // LICM keeps their 16 fragments (128 VGPRs) live across the loop.
  const v16h aw3_0 = wmma_a_frag_lds(W2p, n, PITCH2, 0, lane);
  const v16h aw3_1 = wmma_a_frag_lds(W2p, n, PITCH2, 1, lane);
  const float w2b0 = W2_b[0], w2b1 = W2_b[1];

  const int numIter    = (E + 31) >> 5;               // 32 edges per wave-iteration
  const int wavesTotal = (int)((gridDim.x * blockDim.x) >> 5);
  int it = (int)((blockIdx.x * blockDim.x + tid) >> 5);

  int sCur[2], dCur[2];
  if (it < numIter) load_idx(src, dst, it, n, E, sCur, dCur);

  unsigned lz = 0;   // laundered zero: keeps LAYER-1 weight/bias loads in-loop

  for (; it < numIter; it += wavesTotal) {
    const int base = it * 32;

    asm volatile("" : "+s"(lz));
    const _Float16* winP = Win16 + lz;
    const float*    wbP  = Win_b + lz;

    // ---- software pipeline: next iteration's indices + row prefetch ----
    const int itN = it + wavesTotal;
    int sNext[2], dNext[2];
    if (itN < numIter) {
      load_idx(src, dst, itN, n, E, sNext, dNext);
      if (USE16) {
        __builtin_prefetch(h16 + (size_t)sNext[0] * 64 + 16 * khi, 0, 3);
        __builtin_prefetch(h16 + (size_t)dNext[0] * 64 + 16 * khi, 0, 3);
        __builtin_prefetch(h16 + (size_t)sNext[1] * 64 + 16 * khi, 0, 3);
        __builtin_prefetch(h16 + (size_t)dNext[1] * 64 + 16 * khi, 0, 3);
      } else {
        __builtin_prefetch(h + (size_t)sNext[0] * 64 + 16 * khi, 0, 3);
        __builtin_prefetch(h + (size_t)dNext[0] * 64 + 16 * khi, 0, 3);
        __builtin_prefetch(h + (size_t)sNext[1] * 64 + 16 * khi, 0, 3);
        __builtin_prefetch(h + (size_t)dNext[1] * 64 + 16 * khi, 0, 3);
      }
    } else {
      sNext[0] = sCur[0]; sNext[1] = sCur[1];
      dNext[0] = dCur[0]; dNext[1] = dCur[1];
    }

    // ---- gather both 16-edge tiles as B fragments ----
    v16h hb0[4], hb1[4];
    if (USE16) {
      gather_heB_f16(h16, sCur[0], dCur[0], khi, hb0);
      gather_heB_f16(h16, sCur[1], dCur[1], khi, hb1);
    } else {
      gather_heB_f32(h, sCur[0], dCur[0], khi, hb0);
      gather_heB_f32(h, sCur[1], dCur[1], khi, hb1);
    }

    // ---- layer 1 (8 o-tiles x 4 K-blocks), rolling build of layer-2 B frags ----
    v16h b2_0[4], b2_1[4];
#pragma unroll
    for (int k2 = 0; k2 < 4; ++k2) {
      int pk0[2][4], pk1[2][4];
#pragma unroll
      for (int gg = 0; gg < 2; ++gg) {
        const int g = 2 * k2 + gg;
        const v8f bc = load_bias8(wbP + 16 * g + 8 * khi);
        v8f acc0 = bc, acc1 = bc;
#pragma unroll
        for (int ks = 0; ks < 4; ++ks) {
          v16h aw = wmma_a_frag_lds(winP, 16 * g + n, PITCH1, ks, lane);
          acc0 = __builtin_amdgcn_wmma_f32_16x16x32_f16(false, aw, false, hb0[ks],
                                                        (short)0, acc0, false, false);
          acc1 = __builtin_amdgcn_wmma_f32_16x16x32_f16(false, aw, false, hb1[ks],
                                                        (short)0, acc1, false, false);
        }
        pack_relu(acc0, pk0[gg]);
        pack_relu(acc1, pk1[gg]);
      }
      b2_0[k2] = build_bfrag_swap(pk0[0], pk0[1], khi);
      b2_1[k2] = build_bfrag_swap(pk1[0], pk1[1], khi);
    }

    // ---- layer 2 (4 o-tiles x 4 K-blocks); weight frags stay in registers ----
    v16h b3_0[2], b3_1[2];
#pragma unroll
    for (int k3 = 0; k3 < 2; ++k3) {
      int pk0[2][4], pk1[2][4];
#pragma unroll
      for (int gg = 0; gg < 2; ++gg) {
        const int g = 2 * k3 + gg;
        const v8f bc = load_bias8(W1_b + 16 * g + 8 * khi);
        v8f acc0 = bc, acc1 = bc;
#pragma unroll
        for (int ks = 0; ks < 4; ++ks) {
          v16h aw = wmma_a_frag_lds(W116, 16 * g + n, PITCH1, ks, lane);
          acc0 = __builtin_amdgcn_wmma_f32_16x16x32_f16(false, aw, false, b2_0[ks],
                                                        (short)0, acc0, false, false);
          acc1 = __builtin_amdgcn_wmma_f32_16x16x32_f16(false, aw, false, b2_1[ks],
                                                        (short)0, acc1, false, false);
        }
        pack_relu(acc0, pk0[gg]);
        pack_relu(acc1, pk1[gg]);
      }
      b3_0[k3] = build_bfrag_swap(pk0[0], pk0[1], khi);
      b3_1[k3] = build_bfrag_swap(pk1[0], pk1[1], khi);
    }

    // ---- layer 3: single padded o-tile, 2 K-blocks (weight frags hoisted) ----
    v8f acc0 = {}, acc1 = {};
    acc0 = __builtin_amdgcn_wmma_f32_16x16x32_f16(false, aw3_0, false, b3_0[0],
                                                  (short)0, acc0, false, false);
    acc1 = __builtin_amdgcn_wmma_f32_16x16x32_f16(false, aw3_0, false, b3_1[0],
                                                  (short)0, acc1, false, false);
    acc0 = __builtin_amdgcn_wmma_f32_16x16x32_f16(false, aw3_1, false, b3_0[1],
                                                  (short)0, acc0, false, false);
    acc1 = __builtin_amdgcn_wmma_f32_16x16x32_f16(false, aw3_1, false, b3_1[1],
                                                  (short)0, acc1, false, false);

    // ---- store: lanes 0-15 hold scores (o=0,1) = regs 0,1; one b64 per edge ----
    if (khi == 0) {
      const int e0 = base + n;
      const int e1 = base + 16 + n;
      v2f s0; s0[0] = acc0[0] + w2b0; s0[1] = acc0[1] + w2b1;
      v2f s1; s1[0] = acc1[0] + w2b0; s1[1] = acc1[1] + w2b1;
      if (e0 < E) *(v2f*)(out + 2 * (size_t)e0) = s0;
      if (e1 < E) *(v2f*)(out + 2 * (size_t)e1) = s1;
    }

    sCur[0] = sNext[0]; sCur[1] = sNext[1];
    dCur[0] = dNext[0]; dCur[1] = dNext[1];
  }
}

extern "C" void kernel_launch(void* const* d_in, const int* in_sizes, int n_in,
                              void* d_out, int out_size, void* d_ws, size_t ws_size,
                              hipStream_t stream) {
  (void)n_in; (void)out_size;
  const float* h    = (const float*)d_in[0];
  const int*   src  = (const int*)d_in[1];
  const int*   dst  = (const int*)d_in[2];
  const float* Winw = (const float*)d_in[3];
  const float* Winb = (const float*)d_in[4];
  const float* W1w  = (const float*)d_in[5];
  const float* W1b  = (const float*)d_in[6];
  const float* W2w  = (const float*)d_in[7];
  const float* W2b  = (const float*)d_in[8];
  float* out = (float*)d_out;
  const int E = in_sizes[1];
  const int hTotal = in_sizes[0];               // N * 64 floats

  const size_t shmem = (size_t)(128 * PITCH1 + 64 * PITCH1 + 16 * PITCH2) * 2u;
  const int blocks = 1024;  // 8192 waves over 31250 32-edge iterations

  if (ws_size >= (size_t)hTotal * 2u) {
    _Float16* h16 = (_Float16*)d_ws;
    const int t8 = hTotal / 8;
    h_to_f16<<<(t8 + 255) / 256, 256, 0, stream>>>(h, h16, t8);
    (void)hipFuncSetAttribute(reinterpret_cast<const void*>(&edge_mlp_wmma<true>),
                              hipFuncAttributeMaxDynamicSharedMemorySize, (int)shmem);
    edge_mlp_wmma<true><<<blocks, THREADS, shmem, stream>>>(
        h, h16, src, dst, Winw, Winb, W1w, W1b, W2w, W2b, out, E);
  } else {
    (void)hipFuncSetAttribute(reinterpret_cast<const void*>(&edge_mlp_wmma<false>),
                              hipFuncAttributeMaxDynamicSharedMemorySize, (int)shmem);
    edge_mlp_wmma<false><<<blocks, THREADS, shmem, stream>>>(
        h, (const _Float16*)nullptr, src, dst, Winw, Winb, W1w, W1b, W2w, W2b, out, E);
  }
}
